// _CausalSelfAttention_14542759264533
// MI455X (gfx1250) — matrix operand order repaired and checked by reading, not compiled, not run on hardware
//
#include <hip/hip_runtime.h>

// ---------------------------------------------------------------------------
// MI455X (gfx1250) causal self-attention, bf16 WMMA pipeline.
//   x(2,2048,1024) f32 -> QKV gemm -> flash attention -> out proj -> f32 out
// Matrix math: v_wmma_f32_16x16x32_bf16 (wave32, fp32 accum).
// Data movement: global_load_async_to_lds_b128 (ASYNCcnt) for non-transposed
// tiles; register path only where a transpose is needed (W, V tiles).
// ---------------------------------------------------------------------------

typedef __attribute__((ext_vector_type(16))) __bf16        v16bf;
typedef __attribute__((ext_vector_type(8)))  float         v8f;
typedef __attribute__((ext_vector_type(8)))  unsigned int  v8u;

__device__ __forceinline__ unsigned short f32_to_bf16_rne(float f) {
  unsigned int u = __builtin_bit_cast(unsigned int, f);
  unsigned int r = 0x7FFFu + ((u >> 16) & 1u);
  return (unsigned short)((u + r) >> 16);
}

// Generic LDS pointer -> 32-bit LDS byte offset (low 32 bits per ISA aperture)
__device__ __forceinline__ unsigned int lds_off32(const void* p) {
  return (unsigned int)(unsigned long long)p;
}

// Async DMA: 16 bytes per lane, global -> LDS. Tracked by ASYNCcnt.
__device__ __forceinline__ void async_copy_b128(unsigned int lds_byte_off,
                                                const void* gptr) {
  asm volatile("global_load_async_to_lds_b128 %0, %1, off"
               :: "v"(lds_byte_off), "v"(gptr) : "memory");
}
__device__ __forceinline__ void wait_async0() {
  asm volatile("s_wait_asynccnt 0" ::: "memory");
}

// --- CDNA5 WMMA fragment loaders (wave32 layouts from CDNA5 ISA) --
// A (16x32 bf16): lane L -> M=L&15, h=L>>4 ; VGPR v -> K = 2*(v&3)+16*(v>>2)+8*h
__device__ __forceinline__ v16bf load_fragA(const unsigned short* base, int row,
                                            int strideU16, int kOff, int h) {
  const unsigned int* p = (const unsigned int*)(base + row * strideU16 + kOff);
  v8u u;
#pragma unroll
  for (int v = 0; v < 8; ++v) {
    int k = ((v & 3) << 1) + ((v >> 2) << 4) + (h << 3);
    u[v] = p[k >> 1];
  }
  return __builtin_bit_cast(v16bf, u);
}
// B (32x16 bf16): lane L -> N=L&15, h=L>>4 ; VGPR v -> K = 2*(v&3)+16*(v>>2)+8*h
// Requires LDS layout [N][K] (K contiguous) so each pair is one dword.
__device__ __forceinline__ v16bf load_fragB(const unsigned short* base, int col,
                                            int strideU16, int kOff, int h) {
  const unsigned int* p = (const unsigned int*)(base + col * strideU16 + kOff);
  v8u u;
#pragma unroll
  for (int v = 0; v < 8; ++v) {
    int k = ((v & 3) << 1) + ((v >> 2) << 4) + (h << 3);
    u[v] = p[k >> 1];
  }
  return __builtin_bit_cast(v16bf, u);
}

// ---------------------------------------------------------------------------
// fp32 -> bf16 conversion (4 elems/thread, n % 1024 == 0 for all our tensors)
// ---------------------------------------------------------------------------
__global__ __launch_bounds__(256) void cvt_f32_bf16(const float* __restrict__ in,
                                                    unsigned short* __restrict__ out,
                                                    int n) {
  int i = (blockIdx.x * 256 + threadIdx.x) * 4;
  if (i + 3 < n) {
    float4 f = *(const float4*)(in + i);
    unsigned int lo = (unsigned int)f32_to_bf16_rne(f.x) |
                      ((unsigned int)f32_to_bf16_rne(f.y) << 16);
    unsigned int hi = (unsigned int)f32_to_bf16_rne(f.z) |
                      ((unsigned int)f32_to_bf16_rne(f.w) << 16);
    *(uint2*)(out + i) = make_uint2(lo, hi);
  }
}

// ---------------------------------------------------------------------------
// GEMM + bias: C[M,N] = A_bf16[M,K] @ W_bf16[K,N] + bias ; C is bf16 or f32
// Block tile 128x64, 256 threads = 8 waves (4x2), wave tile 32x32, K step 32.
// A tile streamed via async-to-LDS DMA; W tile transposed through registers.
// ---------------------------------------------------------------------------
#define GA_STR 40  // shA row stride in ushorts: 80B (16B-aligned, conflict-free)

template <bool OUT_BF16>
__global__ __launch_bounds__(256)
void gemm_bias_kernel(const unsigned short* __restrict__ A,
                      const unsigned short* __restrict__ W,
                      const float* __restrict__ bias,
                      void* __restrict__ Cout, int M, int N, int K) {
  __shared__ unsigned short shA[128 * GA_STR]; // [m][k]
  __shared__ unsigned short shB[64 * 34];      // [n][k] (transposed at store)

  const int tid  = threadIdx.x;
  const int lane = tid & 31;
  const int wid  = tid >> 5;
  const int wm   = wid >> 1;           // 0..3 -> 32-row slab
  const int wn   = wid & 1;            // 0..1 -> 32-col slab
  const int h    = (lane >> 4) & 1;
  const int ln   = lane & 15;

  const int m0 = blockIdx.y * 128;
  const int n0 = blockIdx.x * 64;

  const v8f vzero = {0.f, 0.f, 0.f, 0.f, 0.f, 0.f, 0.f, 0.f};
  v8f acc[2][2];
#pragma unroll
  for (int tm = 0; tm < 2; ++tm)
#pragma unroll
    for (int tn = 0; tn < 2; ++tn) acc[tm][tn] = vzero;

  const int arow = tid >> 1, aseg = tid & 1;       // A: 16 bf16 per thread
  const int bk   = tid >> 3, bn   = (tid & 7) * 8; // B: 8 bf16 per thread
  const unsigned int shA_base = lds_off32(shA);

  for (int kb = 0; kb < K; kb += 32) {
    // --- A tile 128x32: async DMA global -> LDS (2x b128 per thread) ---
    {
      const unsigned short* g = A + (size_t)(m0 + arow) * K + kb + aseg * 16;
      const unsigned int l = shA_base + (unsigned)(arow * (GA_STR * 2) + aseg * 32);
      async_copy_b128(l,      g);
      async_copy_b128(l + 16, g + 8);
    }
    // --- B tile 32x64 transposed to [n][k] through registers ---
    {
      uint4 d = *(const uint4*)(W + (size_t)(kb + bk) * N + n0 + bn);
      unsigned int t4[4] = {d.x, d.y, d.z, d.w};
      const unsigned short* s = (const unsigned short*)t4;
#pragma unroll
      for (int i = 0; i < 8; ++i) shB[(bn + i) * 34 + bk] = s[i];
    }
    if (kb + 32 < K) // pull next W tile toward the WGP (global_prefetch_b8)
      __builtin_prefetch(W + (size_t)(kb + 32 + bk) * N + n0 + bn, 0, 0);

    wait_async0();     // my async LDS writes retired
    __syncthreads();   // visible to all waves

    v16bf aF[2], bF[2];
#pragma unroll
    for (int t = 0; t < 2; ++t) aF[t] = load_fragA(shA, wm * 32 + t * 16 + ln, GA_STR, 0, h);
#pragma unroll
    for (int t = 0; t < 2; ++t) bF[t] = load_fragB(shB, wn * 32 + t * 16 + ln, 34, 0, h);
#pragma unroll
    for (int tm = 0; tm < 2; ++tm)
#pragma unroll
      for (int tn = 0; tn < 2; ++tn)
        acc[tm][tn] = __builtin_amdgcn_wmma_f32_16x16x32_bf16(
            false, aF[tm], false, bF[tn], (short)0, acc[tm][tn], false, false);
    __syncthreads();
  }

  // epilogue: C/D layout -> lane N=ln (16-lane half h adds 8 to M), VGPR r -> M
#pragma unroll
  for (int tm = 0; tm < 2; ++tm)
#pragma unroll
    for (int tn = 0; tn < 2; ++tn) {
      const int col = n0 + wn * 32 + tn * 16 + ln;
      const float bv = bias[col];
#pragma unroll
      for (int r = 0; r < 8; ++r) {
        const int row = m0 + wm * 32 + tm * 16 + r + h * 8;
        const float v = acc[tm][tn][r] + bv;
        if (OUT_BF16)
          ((unsigned short*)Cout)[(size_t)row * N + col] = f32_to_bf16_rne(v);
        else
          ((float*)Cout)[(size_t)row * N + col] = v;
      }
    }
}

// ---------------------------------------------------------------------------
// Flash attention: one 64-query tile per block, 4 waves x 16 rows, HD=64.
// qkv bf16 layout: row (b*S+s), cols [0,1024)=Q [1024,2048)=K [2048,3072)=V,
// head hd occupies cols hd*64..hd*64+63 in each slab (matches reshape order).
// Q and K tiles via async-to-LDS DMA; V transposed through registers.
// ---------------------------------------------------------------------------
#define AT_STR 72  // tile row stride in ushorts: 144B (16B-aligned, conflict-free)

__global__ __launch_bounds__(128)
void flash_attn_kernel(const unsigned short* __restrict__ qkv,
                       unsigned short* __restrict__ Obf, int S) {
  __shared__ unsigned short shQ[64 * AT_STR];  // [q][d]
  __shared__ unsigned short shK[64 * AT_STR];  // [k][d]  (B-frag for Q@K^T)
  __shared__ unsigned short shVt[64 * AT_STR]; // [d][k]  (B-frag for P@V)
  __shared__ unsigned short shP[64 * AT_STR];  // [q][k]  (A-frag for P@V)

  const int tid  = threadIdx.x;
  const int lane = tid & 31;
  const int w    = tid >> 5;  // wave -> query rows w*16..w*16+15
  const int h    = (lane >> 4) & 1;
  const int ln   = lane & 15;

  const int bh = blockIdx.y;       // b*NH + head
  const int b  = bh >> 4;
  const int hd = bh & 15;
  const int qt = blockIdx.x;       // query tile index

  const size_t rowQ = (size_t)b * S + (size_t)qt * 64;
  const int    qcol = hd * 64;
  const int    r2 = tid >> 1, seg = tid & 1;

  // issue async Q tile load (each thread: 32 bf16 = 4x b128)
  {
    const unsigned short* g = qkv + (rowQ + r2) * 3072 + qcol + seg * 32;
    const unsigned int l = lds_off32(shQ) + (unsigned)(r2 * (AT_STR * 2) + seg * 64);
#pragma unroll
    for (int i = 0; i < 4; ++i) async_copy_b128(l + i * 16, g + i * 8);
  }

  const v8f vzero = {0.f, 0.f, 0.f, 0.f, 0.f, 0.f, 0.f, 0.f};
  v8f o[4];
#pragma unroll
  for (int dn = 0; dn < 4; ++dn) o[dn] = vzero;
  float mrun[8], lrun[8];
#pragma unroll
  for (int r = 0; r < 8; ++r) { mrun[r] = -1e30f; lrun[r] = 0.f; }

  const float scale = 0.125f; // 1/sqrt(64)

  for (int j = 0; j <= qt; ++j) {
    __syncthreads(); // previous tile fully consumed by all waves
    { // K tile [k][d]: async DMA
      const unsigned short* g =
          qkv + ((size_t)b * S + j * 64 + r2) * 3072 + 1024 + qcol + seg * 32;
      const unsigned int l = lds_off32(shK) + (unsigned)(r2 * (AT_STR * 2) + seg * 64);
#pragma unroll
      for (int i = 0; i < 4; ++i) async_copy_b128(l + i * 16, g + i * 8);
    }
    { // V tile transposed -> shVt[d][k] (register path, 16-bit scatter)
      const uint4* src = (const uint4*)(qkv + ((size_t)b * S + j * 64 + r2) * 3072 +
                                        2048 + qcol + seg * 32);
      unsigned int t16[16];
#pragma unroll
      for (int i = 0; i < 4; ++i) {
        uint4 d = src[i];
        t16[i * 4 + 0] = d.x; t16[i * 4 + 1] = d.y;
        t16[i * 4 + 2] = d.z; t16[i * 4 + 3] = d.w;
      }
      const unsigned short* s = (const unsigned short*)t16;
#pragma unroll
      for (int i = 0; i < 32; ++i) shVt[(seg * 32 + i) * AT_STR + r2] = s[i];
    }
    wait_async0();   // K (and Q on first pass) DMA retired
    __syncthreads(); // all staging visible

    // ---- scores S = Q @ K^T : 8 WMMAs per wave ----
    v8f sc[4];
#pragma unroll
    for (int kn = 0; kn < 4; ++kn) sc[kn] = vzero;
#pragma unroll
    for (int kc = 0; kc < 2; ++kc) {
      v16bf aQ = load_fragA(shQ, w * 16 + ln, AT_STR, kc * 32, h);
#pragma unroll
      for (int kn = 0; kn < 4; ++kn) {
        v16bf bK = load_fragB(shK, kn * 16 + ln, AT_STR, kc * 32, h);
        sc[kn] = __builtin_amdgcn_wmma_f32_16x16x32_bf16(
            false, aQ, false, bK, (short)0, sc[kn], false, false);
      }
    }

    // ---- scale + causal mask (diagonal tile only) ----
    const bool diag = (j == qt);
#pragma unroll
    for (int kn = 0; kn < 4; ++kn)
#pragma unroll
      for (int r = 0; r < 8; ++r) {
        float v = sc[kn][r] * scale;
        if (diag && (kn * 16 + ln > w * 16 + r + h * 8)) v = -1e30f;
        sc[kn][r] = v;
      }

    // ---- online softmax (rows live in 16-lane halves; xor<16 stays in half) --
    float nm[8], al[8], ps[8];
#pragma unroll
    for (int r = 0; r < 8; ++r) {
      float mv = mrun[r];
#pragma unroll
      for (int kn = 0; kn < 4; ++kn) mv = fmaxf(mv, sc[kn][r]);
#pragma unroll
      for (int off = 1; off < 16; off <<= 1) mv = fmaxf(mv, __shfl_xor(mv, off, 32));
      nm[r]   = mv;
      al[r]   = __expf(mrun[r] - mv);
      mrun[r] = mv;
      ps[r]   = 0.f;
    }
#pragma unroll
    for (int kn = 0; kn < 4; ++kn)
#pragma unroll
      for (int r = 0; r < 8; ++r) {
        float p = __expf(sc[kn][r] - nm[r]);
        ps[r] += p;
        // shP region rows w*16..w*16+15 are wave-private; DScnt orders RAW
        shP[(w * 16 + r + h * 8) * AT_STR + kn * 16 + ln] = f32_to_bf16_rne(p);
      }
#pragma unroll
    for (int r = 0; r < 8; ++r) {
      float s = ps[r];
#pragma unroll
      for (int off = 1; off < 16; off <<= 1) s += __shfl_xor(s, off, 32);
      lrun[r] = lrun[r] * al[r] + s;
    }
#pragma unroll
    for (int dn = 0; dn < 4; ++dn)
#pragma unroll
      for (int r = 0; r < 8; ++r) o[dn][r] *= al[r];

    // ---- O += P @ V : 8 WMMAs per wave ----
#pragma unroll
    for (int kc = 0; kc < 2; ++kc) {
      v16bf aP = load_fragA(shP, w * 16 + ln, AT_STR, kc * 32, h);
#pragma unroll
      for (int dn = 0; dn < 4; ++dn) {
        v16bf bV = load_fragB(shVt, dn * 16 + ln, AT_STR, kc * 32, h);
        o[dn] = __builtin_amdgcn_wmma_f32_16x16x32_bf16(
            false, aP, false, bV, (short)0, o[dn], false, false);
      }
    }
  }

  // ---- normalize + store bf16 O[b*S+s][hd*64+d] ----
#pragma unroll
  for (int dn = 0; dn < 4; ++dn)
#pragma unroll
    for (int r = 0; r < 8; ++r) {
      const size_t row = rowQ + (size_t)(w * 16 + r + h * 8);
      const int    col = qcol + dn * 16 + ln;
      Obf[row * 1024 + col] = f32_to_bf16_rne(o[dn][r] / lrun[r]);
    }
}

// ---------------------------------------------------------------------------
extern "C" void kernel_launch(void* const* d_in, const int* in_sizes, int n_in,
                              void* d_out, int out_size, void* d_ws, size_t ws_size,
                              hipStream_t stream) {
  (void)in_sizes; (void)n_in; (void)out_size; (void)ws_size;
  const float* x     = (const float*)d_in[0];
  const float* w_qkv = (const float*)d_in[1];
  const float* b_qkv = (const float*)d_in[2];
  const float* w_out = (const float*)d_in[3];
  const float* b_out = (const float*)d_in[4];
  float* out = (float*)d_out;

  const int B = 2, S = 2048, H = 1024;
  const int M  = B * S;   // 4096
  const int N3 = 3 * H;   // 3072

  char* ws = (char*)d_ws;
  unsigned short* x_bf    = (unsigned short*)ws; ws += (size_t)M * H * 2;
  unsigned short* wqkv_bf = (unsigned short*)ws; ws += (size_t)H * N3 * 2;
  unsigned short* wout_bf = (unsigned short*)ws; ws += (size_t)H * H * 2;
  unsigned short* qkv_bf  = (unsigned short*)ws; ws += (size_t)M * N3 * 2;
  unsigned short* o_bf    = (unsigned short*)ws; ws += (size_t)M * H * 2;

  cvt_f32_bf16<<<(M * H) / 1024, 256, 0, stream>>>(x, x_bf, M * H);
  cvt_f32_bf16<<<(H * N3) / 1024, 256, 0, stream>>>(w_qkv, wqkv_bf, H * N3);
  cvt_f32_bf16<<<(H * H) / 1024, 256, 0, stream>>>(w_out, wout_bf, H * H);

  gemm_bias_kernel<true><<<dim3(N3 / 64, M / 128), 256, 0, stream>>>(
      x_bf, wqkv_bf, b_qkv, qkv_bf, M, N3, H);

  flash_attn_kernel<<<dim3(S / 64, B * 16), 128, 0, stream>>>(qkv_bf, o_bf, S);

  gemm_bias_kernel<false><<<dim3(H / 64, M / 128), 256, 0, stream>>>(
      o_bf, wout_bf, b_out, out, M, H, H);
}
